// MultiheadAttention_4569845203240
// MI455X (gfx1250) — matrix operand order repaired and checked by reading, not compiled, not run on hardware
//
#include <hip/hip_runtime.h>
#include <hip/hip_bf16.h>

// ---------------------------------------------------------------------------
// MHA forward for MI455X (gfx1250), wave32 + v_wmma_f32_16x16x32_bf16.
//   B=4, L=S=2048, D=1024, H=16, dh=64, causal softmax.
// Pipeline:
//   1) cvt fp32->bf16: 4 weights + 3 activations (bandwidth pass)
//   2) unified bf16 GEMM y = x*W^T (32x64 wave tile, 8 WMMA / k-step)
//        mode 0/1: per-head [B,H,T,64]   mode 2: V^T [B,H,64,S]
//        mode 3: fp32 output (final projection)
//   3) causal flash attention (WMMA scores + WMMA P*V, LDS transpose)
// ---------------------------------------------------------------------------

#define BATCH 4
#define SEQL  2048
#define SEQS  2048
#define DIM   1024
#define NHEAD 16
#define DH    64

typedef float v8f  __attribute__((ext_vector_type(8)));
typedef __bf16 v16bf __attribute__((ext_vector_type(16)));

union FragAB {              // 16 bf16 = 32 bytes = 8 VGPRs (A or B operand)
    v16bf v;
    uint4 q[2];
};

__device__ __forceinline__ unsigned short f2bf(float x) {
    unsigned u = __float_as_uint(x);
    u += 0x7FFFu + ((u >> 16) & 1u);       // round-to-nearest-even
    return (unsigned short)(u >> 16);
}

__device__ __forceinline__ v8f wmma_bf16(const v16bf a, const v16bf b, v8f c) {
    // D = A(16x32) * B(32x16) + C, f32 accumulate
    return __builtin_amdgcn_wmma_f32_16x16x32_bf16(
        false, a, false, b, (short)0, c, false, false);
}

// ---------------------------------------------------------------------------
// Kernel 1: fp32 -> bf16 conversion (vectorized, pure bandwidth).
// ---------------------------------------------------------------------------
__global__ void cvt_f32_bf16_kernel(const float* __restrict__ src,
                                    unsigned short* __restrict__ dst, int n4) {
    int i = blockIdx.x * blockDim.x + threadIdx.x;
    if (i < n4) {
        float4 v = ((const float4*)src)[i];
        uint2 o;
        o.x = (unsigned)f2bf(v.x) | ((unsigned)f2bf(v.y) << 16);
        o.y = (unsigned)f2bf(v.z) | ((unsigned)f2bf(v.w) << 16);
        ((uint2*)dst)[i] = o;
    }
}

// ---------------------------------------------------------------------------
// Kernel 2: unified GEMM  C[M,N] = A[M,K](bf16) * W[N,K](bf16)^T
//   One wave computes a 32(M) x 64(N) tile; K = 1024 in steps of 32.
//   A fragments: lane = row, 16B chunks at k0+sel*8 / k0+16+sel*8.
//   B fragments: lane = col (W row), 16B chunks at k0+sel*8 / k0+16+sel*8.
// ---------------------------------------------------------------------------
__global__ void __launch_bounds__(32)
gemm_xWT_kernel(const unsigned short* __restrict__ Abf,
                const unsigned short* __restrict__ Wb,
                void* __restrict__ OutV, int mode) {
    const int lane = threadIdx.x;
    const int ln   = lane & 15;
    const int half = lane >> 4;
    const int m0   = blockIdx.x * 32;
    const int n0   = blockIdx.y * 64;

    const unsigned short* arow0 = Abf + (size_t)(m0 + ln) * DIM;
    const unsigned short* arow1 = arow0 + (size_t)16 * DIM;
    v8f c[2][4] = {};

    for (int k0 = 0; k0 < DIM; k0 += 32) {
        FragAB a0, a1;
        a0.q[0] = *(const uint4*)(arow0 + k0 + half * 8);
        a0.q[1] = *(const uint4*)(arow0 + k0 + 16 + half * 8);
        a1.q[0] = *(const uint4*)(arow1 + k0 + half * 8);
        a1.q[1] = *(const uint4*)(arow1 + k0 + 16 + half * 8);
#pragma unroll
        for (int j = 0; j < 4; ++j) {
            const unsigned short* wr =
                Wb + (size_t)(n0 + j * 16 + ln) * DIM + k0 + half * 8;
            FragAB b;
            b.q[0] = *(const uint4*)wr;
            b.q[1] = *(const uint4*)(wr + 16);
            c[0][j] = wmma_bf16(a0.v, b.v, c[0][j]);
            c[1][j] = wmma_bf16(a1.v, b.v, c[1][j]);
        }
    }

#pragma unroll
    for (int i = 0; i < 2; ++i) {
#pragma unroll
        for (int r = 0; r < 8; ++r) {              // C: lane=col, VGPR r = row (+8 upper half)
            int m  = m0 + i * 16 + r + 8 * half;
            int bb = m >> 11;                      // / SEQL
            int t  = m & (SEQL - 1);
#pragma unroll
            for (int j = 0; j < 4; ++j) {
                int n = n0 + j * 16 + ln;
                float val = c[i][j][r];
                if (mode == 3) {                   // final projection: fp32 [B*L, D]
                    ((float*)OutV)[(size_t)m * DIM + n] = val;
                } else {
                    int h = n >> 6, d = n & 63;
                    size_t idx;
                    if (mode == 2)                 // V^T: [B,H,DH,S]
                        idx = (((size_t)bb * NHEAD + h) * DH + d) * SEQS + t;
                    else                           // Q/K: [B,H,T,DH]
                        idx = (((size_t)bb * NHEAD + h) * SEQL + t) * DH + d;
                    ((unsigned short*)OutV)[idx] = f2bf(val);
                }
            }
        }
    }
}

// ---------------------------------------------------------------------------
// Kernel 3: causal flash attention, one wave per 16-row Q tile per (b,h).
//   Scores and P*V on WMMA; online softmax via half-wave shuffles;
//   P transposed C-layout -> A-layout through LDS (same-wave DS is in-order).
// ---------------------------------------------------------------------------
__global__ void __launch_bounds__(32)
flash_attn_kernel(const unsigned short* __restrict__ Qh,
                  const unsigned short* __restrict__ Kh,
                  const unsigned short* __restrict__ VT,
                  unsigned short* __restrict__ Oa) {
    __shared__ unsigned short pls[16][40];         // 16x32 P tile, padded rows (80B, 16B-aligned)

    const int lane = threadIdx.x;
    const int ln   = lane & 15;
    const int half = lane >> 4;
    const int l0   = blockIdx.x * 16;
    const int h    = blockIdx.y;
    const int b    = blockIdx.z;

    const float KSC = 0.125f * 1.44269504088896f;  // log2(e)/sqrt(dh)
    const float NEG_INF = -__builtin_inff();

    // Q tile as two A fragments (dh = 64 = 2 x 32-k steps)
    const unsigned short* qbase =
        Qh + (((size_t)b * NHEAD + h) * SEQL + (l0 + ln)) * DH;
    FragAB aq0, aq1;
    aq0.q[0] = *(const uint4*)(qbase + half * 8);
    aq0.q[1] = *(const uint4*)(qbase + 16 + half * 8);
    aq1.q[0] = *(const uint4*)(qbase + 32 + half * 8);
    aq1.q[1] = *(const uint4*)(qbase + 48 + half * 8);

    const unsigned short* kbase  = Kh + ((size_t)b * NHEAD + h) * SEQS * DH;
    const unsigned short* vtbase = VT + ((size_t)b * NHEAD + h) * DH * SEQS;

    float mrun[8], lrun[8];
#pragma unroll
    for (int r = 0; r < 8; ++r) { mrun[r] = NEG_INF; lrun[r] = 0.0f; }
    v8f oacc[4] = {};

    for (int s0 = 0; s0 < l0 + 16; s0 += 32) {
        // ---- scores: two 16x16 tiles (s columns s0..15, s0+16..31) ----
        v8f c0 = {}, c1 = {};
        {
            const unsigned short* kr = kbase + (size_t)(s0 + ln) * DH;
            FragAB b0, b1;                          // B = K^T: lane=s col, k=d chunks at 8h / 16+8h
            b0.q[0] = *(const uint4*)(kr + half * 8);
            b0.q[1] = *(const uint4*)(kr + 16 + half * 8);
            b1.q[0] = *(const uint4*)(kr + 32 + half * 8);
            b1.q[1] = *(const uint4*)(kr + 48 + half * 8);
            c0 = wmma_bf16(aq0.v, b0.v, c0);
            c0 = wmma_bf16(aq1.v, b1.v, c0);
        }
        {
            const unsigned short* kr = kbase + (size_t)(s0 + 16 + ln) * DH;
            FragAB b0, b1;
            b0.q[0] = *(const uint4*)(kr + half * 8);
            b0.q[1] = *(const uint4*)(kr + 16 + half * 8);
            b1.q[0] = *(const uint4*)(kr + 32 + half * 8);
            b1.q[1] = *(const uint4*)(kr + 48 + half * 8);
            c1 = wmma_bf16(aq0.v, b0.v, c1);
            c1 = wmma_bf16(aq1.v, b1.v, c1);
        }

        // ---- causal mask (only diagonal-straddling chunks) ----
        if (s0 + 31 > l0 + 15) {
#pragma unroll
            for (int r = 0; r < 8; ++r) {
                int lrow = l0 + r + 8 * half;
                if (s0 + ln > lrow)      c0[r] = NEG_INF;
                if (s0 + 16 + ln > lrow) c1[r] = NEG_INF;
            }
        }

        // ---- online softmax (row stats via half-wave xor reductions) ----
        float p0[8], p1[8];
#pragma unroll
        for (int r = 0; r < 8; ++r) {
            float pm = fmaxf(c0[r], c1[r]);
            pm = fmaxf(pm, __shfl_xor(pm, 1, 32));
            pm = fmaxf(pm, __shfl_xor(pm, 2, 32));
            pm = fmaxf(pm, __shfl_xor(pm, 4, 32));
            pm = fmaxf(pm, __shfl_xor(pm, 8, 32));
            float mnew  = fmaxf(mrun[r], pm);
            float alpha = exp2f((mrun[r] - mnew) * KSC);
            p0[r] = exp2f((c0[r] - mnew) * KSC);
            p1[r] = exp2f((c1[r] - mnew) * KSC);
            float rs = p0[r] + p1[r];
            rs += __shfl_xor(rs, 1, 32);
            rs += __shfl_xor(rs, 2, 32);
            rs += __shfl_xor(rs, 4, 32);
            rs += __shfl_xor(rs, 8, 32);
            lrun[r] = lrun[r] * alpha + rs;
            mrun[r] = mnew;
#pragma unroll
            for (int j = 0; j < 4; ++j) oacc[j][r] = oacc[j][r] * alpha;
        }

        // ---- transpose P (C layout -> A layout) through LDS, as bf16 ----
#pragma unroll
        for (int r = 0; r < 8; ++r) {
            int row = r + 8 * half;
            pls[row][ln]      = f2bf(p0[r]);
            pls[row][ln + 16] = f2bf(p1[r]);
        }
        FragAB ap;                                  // A: lane=row(ln), k=s within chunk
        const unsigned short* prow = &pls[ln][0];
        ap.q[0] = *(const uint4*)(prow + half * 8);
        ap.q[1] = *(const uint4*)(prow + 16 + half * 8);

        // ---- O += P(16x32) * V(32x64) using V^T rows (contiguous in s) ----
#pragma unroll
        for (int j = 0; j < 4; ++j) {
            const unsigned short* vr =
                vtbase + (size_t)(j * 16 + ln) * SEQS + s0 + half * 8;
            FragAB bv;
            bv.q[0] = *(const uint4*)vr;
            bv.q[1] = *(const uint4*)(vr + 16);
            oacc[j] = wmma_bf16(ap.v, bv.v, oacc[j]);
        }
    }

    // ---- normalize and write attn output as [B, L, H*DH] bf16 ----
#pragma unroll
    for (int r = 0; r < 8; ++r) {
        float inv = 1.0f / lrun[r];
        int lrow = l0 + r + 8 * half;
        size_t obase = ((size_t)b * SEQL + lrow) * DIM + h * DH;
#pragma unroll
        for (int j = 0; j < 4; ++j)
            Oa[obase + j * 16 + ln] = f2bf(oacc[j][r] * inv);
    }
}

// ---------------------------------------------------------------------------
// Host launcher
// ---------------------------------------------------------------------------
extern "C" void kernel_launch(void* const* d_in, const int* in_sizes, int n_in,
                              void* d_out, int out_size, void* d_ws, size_t ws_size,
                              hipStream_t stream) {
    (void)in_sizes; (void)n_in; (void)out_size; (void)ws_size;

    const float* queries = (const float*)d_in[0];
    const float* keys    = (const float*)d_in[1];
    const float* values  = (const float*)d_in[2];
    const float* Wq      = (const float*)d_in[3];
    const float* Wk      = (const float*)d_in[4];
    const float* Wv      = (const float*)d_in[5];
    const float* Wo      = (const float*)d_in[6];
    float* out           = (float*)d_out;

    char* ws = (char*)d_ws;
    const size_t WBYTES = (size_t)DIM * DIM * 2;                 // 2 MB each
    const size_t TBYTES = (size_t)BATCH * SEQL * DIM * 2;        // 16 MB each
    unsigned short* WqB = (unsigned short*)(ws);
    unsigned short* WkB = (unsigned short*)(ws + WBYTES);
    unsigned short* WvB = (unsigned short*)(ws + 2 * WBYTES);
    unsigned short* WoB = (unsigned short*)(ws + 3 * WBYTES);
    unsigned short* Xq  = (unsigned short*)(ws + 4 * WBYTES);             // bf16 queries
    unsigned short* Xk  = (unsigned short*)(ws + 4 * WBYTES + TBYTES);    // bf16 keys
    unsigned short* Xv  = (unsigned short*)(ws + 4 * WBYTES + 2 * TBYTES);// bf16 values
    unsigned short* Qh  = (unsigned short*)(ws + 4 * WBYTES + 3 * TBYTES);
    unsigned short* Kh  = (unsigned short*)(ws + 4 * WBYTES + 4 * TBYTES);
    unsigned short* VT  = (unsigned short*)(ws + 4 * WBYTES + 5 * TBYTES);
    unsigned short* Oa  = Xq;   // Xq is dead after the Q projection -> reuse

    // 1) fp32 -> bf16: weights and activations (pure bandwidth pass)
    {
        int wn4 = (DIM * DIM) / 4;                    // 262144 float4s
        dim3 wgrid((wn4 + 255) / 256), blk(256);
        cvt_f32_bf16_kernel<<<wgrid, blk, 0, stream>>>(Wq, WqB, wn4);
        cvt_f32_bf16_kernel<<<wgrid, blk, 0, stream>>>(Wk, WkB, wn4);
        cvt_f32_bf16_kernel<<<wgrid, blk, 0, stream>>>(Wv, WvB, wn4);
        cvt_f32_bf16_kernel<<<wgrid, blk, 0, stream>>>(Wo, WoB, wn4);
        int xn4 = (BATCH * SEQL * DIM) / 4;           // 2097152 float4s
        dim3 xgrid((xn4 + 255) / 256);
        cvt_f32_bf16_kernel<<<xgrid, blk, 0, stream>>>(queries, Xq, xn4);
        cvt_f32_bf16_kernel<<<xgrid, blk, 0, stream>>>(keys,    Xk, xn4);
        cvt_f32_bf16_kernel<<<xgrid, blk, 0, stream>>>(values,  Xv, xn4);
    }

    // 2) Q/K/V projections (V stored transposed per head)
    {
        dim3 grid((BATCH * SEQL) / 32, DIM / 64), blk(32);
        gemm_xWT_kernel<<<grid, blk, 0, stream>>>(Xq, WqB, (void*)Qh, 0);
        gemm_xWT_kernel<<<grid, blk, 0, stream>>>(Xk, WkB, (void*)Kh, 1);
        gemm_xWT_kernel<<<grid, blk, 0, stream>>>(Xv, WvB, (void*)VT, 2);
    }

    // 3) causal flash attention (writes Oa, aliased over dead Xq)
    {
        dim3 grid(SEQL / 16, NHEAD, BATCH), blk(32);
        flash_attn_kernel<<<grid, blk, 0, stream>>>(Qh, Kh, VT, Oa);
    }

    // 4) output projection -> fp32
    {
        dim3 grid((BATCH * SEQL) / 32, DIM / 64), blk(32);
        gemm_xWT_kernel<<<grid, blk, 0, stream>>>(Oa, WoB, (void*)out, 3);
    }
}
